// GNNModel_47923245088955
// MI455X (gfx1250) — matrix operand order repaired and checked by reading, not compiled, not run on hardware
//
#include <hip/hip_runtime.h>
#include <hip/hip_bf16.h>
#include <math.h>

// ---------------------------------------------------------------------------
// CDNA5 (gfx1250) GNN implementation.  All dense GEMMs run through a bf16
// V_WMMA_F32_16X16X32_BF16 kernel whose operands are pre-packed + zero-padded
// so the K-loop is branch-free.  Each wave computes a 32x64 tile: 2 A frags
// (2x b128 each) + 4 shared B frags (2x b128 each, K-major) -> 8 WMMAs per
// 32-deep K step per wave (12 b128 loads / 8 WMMAs).
// Graph ops are deterministic gathers (constant degree k+1 -> no atomics).
// Top-k stages a full 8192-float distance row in LDS (32KB << 320KB/WGP).
// ---------------------------------------------------------------------------

typedef __attribute__((ext_vector_type(16))) __bf16 v16bf;
typedef __attribute__((ext_vector_type(8)))  __bf16 v8bf;
typedef __attribute__((ext_vector_type(8)))  float  v8f;

__device__ inline unsigned short f32_to_bf16_bits(float f) {
  unsigned int x = __float_as_uint(f);
  unsigned int r = x + 0x7FFFu + ((x >> 16) & 1u);  // round-to-nearest-even
  return (unsigned short)(r >> 16);
}

// ---- operand packing (pad to Mp x Kp, zero fill) ---------------------------
__global__ void pack_a_bf16(const float* __restrict__ in, unsigned short* __restrict__ out,
                            int rows, int cols, int ldin, int Mp, int Kp) {
  long long idx = (long long)blockIdx.x * 256 + threadIdx.x;
  long long total = (long long)Mp * Kp;
  if (idx >= total) return;
  int i = (int)(idx / Kp);
  int k = (int)(idx % Kp);
  float v = (i < rows && k < cols) ? in[(size_t)i * ldin + k] : 0.f;
  out[idx] = f32_to_bf16_bits(v);
}

// weights [K,N] row-major -> packed K-major [Np, Kp]
__global__ void pack_bT_bf16(const float* __restrict__ w, unsigned short* __restrict__ out,
                             int K, int N, int Np, int Kp) {
  long long idx = (long long)blockIdx.x * 256 + threadIdx.x;
  long long total = (long long)Np * Kp;
  if (idx >= total) return;
  int n = (int)(idx / Kp);
  int k = (int)(idx % Kp);
  float v = (n < N && k < K) ? w[(size_t)k * N + n] : 0.f;
  out[idx] = f32_to_bf16_bits(v);
}

// ------------------------------ WMMA GEMM ----------------------------------
// C[M,N] = act( A_pk[Mp,Kp] @ B_pk[Np,Kp]^T + bias )
// Block = 4 waves; wave w -> rows [by*128 + w*32, +32) x cols [bx*64, +64).
// M must be a multiple of 128 (true for all call sites: 8192/4096/2048).
__global__ void __launch_bounds__(128)
wmma_gemm_bf16(const unsigned short* __restrict__ A,
               const unsigned short* __restrict__ B,
               const float* __restrict__ bias,
               float* __restrict__ C,
               int M, int N, int Kp, int ldc, int act) {
  const int lane = threadIdx.x & 31;
  const int wave = threadIdx.x >> 5;
  const int m0 = blockIdx.y * 128 + wave * 32;
  const int n0 = blockIdx.x * 64;
  const int r  = lane & 15;
  const int hi = lane >> 4;

  const unsigned short* arow0 = A + (size_t)(m0 + r) * Kp + hi * 8;
  const unsigned short* arow1 = arow0 + (size_t)16 * Kp;
  const unsigned short* brow0 = B + (size_t)(n0 + r) * Kp + hi * 8;
  const size_t bstep = (size_t)16 * Kp;   // next 16-column group of B

  v8f acc[2][4];
#pragma unroll
  for (int u = 0; u < 2; ++u)
#pragma unroll
    for (int t = 0; t < 4; ++t)
#pragma unroll
      for (int v = 0; v < 8; ++v) acc[u][t][v] = 0.0f;

  union { v16bf v; v8bf h[2]; } ua0, ua1, ub;

  for (int kk = 0; kk < Kp; kk += 32) {
    if (kk + 64 <= Kp) {                       // uniform-branch prefetch
      __builtin_prefetch(arow0 + kk + 32, 0, 1);
      __builtin_prefetch(arow1 + kk + 32, 0, 1);
      __builtin_prefetch(brow0 + kk + 32, 0, 1);
    }
    // A fragments: two contiguous 8-element (16B) runs per lane, two M tiles
    ua0.h[0] = *(const v8bf*)(arow0 + kk);
    ua0.h[1] = *(const v8bf*)(arow0 + kk + 16);
    ua1.h[0] = *(const v8bf*)(arow1 + kk);
    ua1.h[1] = *(const v8bf*)(arow1 + kk + 16);
    v16bf a0 = ua0.v;
    v16bf a1 = ua1.v;
#pragma unroll
    for (int t = 0; t < 4; ++t) {
      // B fragment: two contiguous 8-element (16B) runs per lane, K-major packed;
      // shared by both M tiles (halves per-FLOP load count)
      const unsigned short* bp = brow0 + (size_t)t * bstep + kk;
      ub.h[0] = *(const v8bf*)(bp);
      ub.h[1] = *(const v8bf*)(bp + 16);
      v16bf b = ub.v;
      acc[0][t] = __builtin_amdgcn_wmma_f32_16x16x32_bf16(
          false, a0, false, b, (short)0, acc[0][t], false, false);
      acc[1][t] = __builtin_amdgcn_wmma_f32_16x16x32_bf16(
          false, a1, false, b, (short)0, acc[1][t], false, false);
    }
  }
  // masked epilogue (out of the hot loop)
#pragma unroll
  for (int u = 0; u < 2; ++u) {
#pragma unroll
    for (int t = 0; t < 4; ++t) {
#pragma unroll
      for (int v = 0; v < 8; ++v) {
        int om = m0 + u * 16 + v + hi * 8;
        int on = n0 + t * 16 + r;
        if (om < M && on < N) {
          float val = acc[u][t][v];
          if (bias) val += bias[on];
          if (act == 1)      val = val > 0.f ? val : 0.f;       // relu
          else if (act == 2) val = 1.f / (1.f + expf(-val));    // sigmoid
          C[(size_t)om * ldc + on] = val;
        }
      }
    }
  }
}

// ------------------------------ kNN helpers --------------------------------
__global__ void sqnorm_kernel(const float* __restrict__ f, float* __restrict__ sq,
                              int n, int dim, int ld) {
  int i = blockIdx.x * 256 + threadIdx.x;
  if (i >= n) return;
  float acc = 0.f;
  for (int c = 0; c < dim; ++c) { float v = f[(size_t)i * ld + c]; acc += v * v; }
  sq[i] = acc;
}

__global__ void dist_finish_kernel(float* __restrict__ G, const float* __restrict__ sq,
                                   int r0, int rows, int n) {
  long long idx = (long long)blockIdx.x * 256 + threadIdx.x;
  long long total = (long long)rows * n;
  if (idx >= total) return;
  int i = (int)(idx / n);
  int j = (int)(idx % n);
  float d = sq[r0 + i] - 2.f * G[idx] + sq[j];
  if (r0 + i == j) d = 3.0e38f;   // loop=False
  G[idx] = d;
}

// One block per row; full row staged in LDS, k serial argmin tree reductions.
__global__ void topk_kernel(const float* __restrict__ Dblk, int r0, int n, int k,
                            int* __restrict__ srcOut) {
  __shared__ float row[8192];
  __shared__ unsigned long long red[256];
  const int i = blockIdx.x;
  const float* dr = Dblk + (size_t)i * n;
  for (int j = threadIdx.x; j < n; j += 256) row[j] = dr[j];
  __syncthreads();
  for (int sel = 0; sel < k; ++sel) {
    unsigned long long best = ~0ULL;
    for (int j = threadIdx.x; j < n; j += 256) {
      unsigned int ub = __float_as_uint(row[j]);
      ub = (ub & 0x80000000u) ? ~ub : (ub | 0x80000000u);   // order-preserving
      unsigned long long key = ((unsigned long long)ub << 32) | (unsigned int)j;
      if (key < best) best = key;
    }
    red[threadIdx.x] = best;
    __syncthreads();
    for (int s = 128; s > 0; s >>= 1) {
      if (threadIdx.x < s && red[threadIdx.x + s] < red[threadIdx.x])
        red[threadIdx.x] = red[threadIdx.x + s];
      __syncthreads();
    }
    int idx = (int)(red[0] & 0xFFFFFFFFu);
    if (threadIdx.x == 0) {
      srcOut[(size_t)(r0 + i) * k + sel] = idx;
      row[idx] = 3.0e38f;
    }
    __syncthreads();
  }
}

// ------------------------------ GCN / GAT ----------------------------------
__global__ void gcn_agg_kernel(const float* __restrict__ xw, const int* __restrict__ src,
                               const float* __restrict__ bias, float* __restrict__ out,
                               int ldout, int n, int c, int k) {
  long long idx = (long long)blockIdx.x * 256 + threadIdx.x;
  if (idx >= (long long)n * c) return;
  int i  = (int)(idx / c);
  int cc = (int)(idx % c);
  float acc = xw[(size_t)i * c + cc];                // self loop
  for (int j = 0; j < k; ++j)
    acc += xw[(size_t)src[(size_t)i * k + j] * c + cc];
  float v = acc / (float)(k + 1) + bias[cc];
  out[(size_t)i * ldout + cc] = v > 0.f ? v : 0.f;   // relu follows every gcn
}

__global__ void gat_scores_kernel(const float* __restrict__ xw,
                                  const float* __restrict__ a_s,
                                  const float* __restrict__ a_d,
                                  float* __restrict__ es, float* __restrict__ ed,
                                  int n, int h, int c) {
  int idx = blockIdx.x * 256 + threadIdx.x;
  if (idx >= n * h) return;
  int i = idx / h, hd = idx % h;
  const float* x = xw + (size_t)i * h * c + (size_t)hd * c;
  float s = 0.f, d = 0.f;
  for (int cc = 0; cc < c; ++cc) {
    s += x[cc] * a_s[hd * c + cc];
    d += x[cc] * a_d[hd * c + cc];
  }
  es[idx] = s;
  ed[idx] = d;
}

__global__ void gat_alpha_kernel(const float* __restrict__ es, const float* __restrict__ ed,
                                 const int* __restrict__ src, float* __restrict__ alpha,
                                 int n, int h, int k) {
  int idx = blockIdx.x * 256 + threadIdx.x;
  if (idx >= n * h) return;
  int i = idx / h, hd = idx % h;
  float edi = ed[i * h + hd];
  float ebuf[24];
  float m = -3.4e38f;
  for (int j = 0; j <= k; ++j) {
    int s = (j < k) ? src[(size_t)i * k + j] : i;
    float e = es[s * h + hd] + edi;
    e = e > 0.f ? e : 0.2f * e;                       // leaky_relu(0.2)
    ebuf[j] = e;
    if (e > m) m = e;
  }
  float den = 0.f;
  for (int j = 0; j <= k; ++j) { float ex = expf(ebuf[j] - m); ebuf[j] = ex; den += ex; }
  float inv = 1.f / den;
  for (int j = 0; j <= k; ++j)
    alpha[((size_t)i * (k + 1) + j) * h + hd] = ebuf[j] * inv;
}

__global__ void gat_out_kernel(const float* __restrict__ xw, const float* __restrict__ alpha,
                               const int* __restrict__ src, const float* __restrict__ bias,
                               float* __restrict__ out, int ldout,
                               int n, int h, int c, int k) {
  long long idx = (long long)blockIdx.x * 256 + threadIdx.x;
  if (idx >= (long long)n * c) return;
  int i  = (int)(idx / c);
  int cc = (int)(idx % c);
  float acc = 0.f;
  for (int j = 0; j <= k; ++j) {
    int s = (j < k) ? src[(size_t)i * k + j] : i;
    const float* xr = xw + (size_t)s * h * c + cc;
    const float* ar = alpha + ((size_t)i * (k + 1) + j) * h;
    for (int hd = 0; hd < h; ++hd) acc += ar[hd] * xr[(size_t)hd * c];
  }
  float v = acc / (float)h + bias[cc];
  out[(size_t)i * ldout + cc] = v > 0.f ? v : 0.f;   // relu follows every gat
}

// --------------------------- edge MLP helpers ------------------------------
__global__ void edge_feat_kernel(const float* __restrict__ xf, const int* __restrict__ src2,
                                 float* __restrict__ feats, int e0, int ce, int k2) {
  long long idx = (long long)blockIdx.x * 256 + threadIdx.x;
  long long total = (long long)ce * 1408;
  if (idx >= total) return;
  int el  = (int)(idx / 1408);
  int col = (int)(idx % 1408);
  int e = e0 + el;
  int s = src2[e];
  int d = e / k2;
  float v = (col < 704) ? xf[(size_t)s * 704 + col]
                        : xf[(size_t)d * 704 + (col - 704)];
  feats[(size_t)el * 1409 + col] = v;
}

__global__ void edge_weight_kernel(const float* __restrict__ xf, const int* __restrict__ src2,
                                   float* __restrict__ feats, int e0, int ce, int k2) {
  int el = blockIdx.x * 256 + threadIdx.x;
  if (el >= ce) return;
  int e = e0 + el;
  const float* a = xf + (size_t)src2[e] * 704;
  const float* b = xf + (size_t)(e / k2) * 704;
  float acc = 0.f;
  for (int c = 0; c < 704; ++c) { float df = a[c] - b[c]; acc += df * df; }
  feats[(size_t)el * 1409 + 1408] = sqrtf(acc);
}

__global__ void labels_kernel(const float* __restrict__ y, const int* __restrict__ src2,
                              float* __restrict__ out, int E, int k2) {
  int e = blockIdx.x * 256 + threadIdx.x;
  if (e >= E) return;
  int s = src2[e], d = e / k2;
  float ys0 = y[s * 3], ys1 = y[s * 3 + 1], ys2 = y[s * 3 + 2];
  float yd0 = y[d * 3], yd1 = y[d * 3 + 1], yd2 = y[d * 3 + 2];
  float ls = ys0 + ys1 + ys2, le = yd0 + yd1 + yd2;
  float d0 = ys0 - yd0, d1 = ys1 - yd1, d2 = ys2 - yd2;
  float diff = sqrtf(d0 * d0 + d1 * d1 + d2 * d2);
  float tot = ls + le;
  float lbl = 0.f;
  if (tot == 2.f)      lbl = (diff == 0.f) ? 0.99f : 0.5f;
  else if (tot == 1.f) lbl = 0.1f;
  else if (tot == 0.f) lbl = 0.01f;
  out[e] = lbl;
}

// ===========================================================================
static inline int rup(int x, int m) { return (x + m - 1) / m * m; }

extern "C" void kernel_launch(void* const* d_in, const int* in_sizes, int n_in,
                              void* d_out, int out_size, void* d_ws, size_t ws_size,
                              hipStream_t stream) {
  const int N = 8192, K1 = 20, K2 = 10;
  const int E2 = N * K2;                 // 81920
  const int CH = 4096;                   // edge-MLP chunk (E2 = 20*CH)

  const float* x = (const float*)d_in[0];
  const float* y = (const float*)d_in[2];
  int p = 3;
  const float* nn1_w1 = (const float*)d_in[p++]; const float* nn1_b1 = (const float*)d_in[p++];
  const float* nn1_w2 = (const float*)d_in[p++]; const float* nn1_b2 = (const float*)d_in[p++];
  const float* nn1_w3 = (const float*)d_in[p++]; const float* nn1_b3 = (const float*)d_in[p++];
  const float* gcn1_w = (const float*)d_in[p++]; const float* gcn1_b = (const float*)d_in[p++];
  const float* gcn2_w = (const float*)d_in[p++]; const float* gcn2_b = (const float*)d_in[p++];
  const float* gcn3_w = (const float*)d_in[p++]; const float* gcn3_b = (const float*)d_in[p++];
  const float* gcn4_w = (const float*)d_in[p++]; const float* gcn4_b = (const float*)d_in[p++];
  const float* gat1_w = (const float*)d_in[p++]; const float* gat1_as = (const float*)d_in[p++];
  const float* gat1_ad = (const float*)d_in[p++]; const float* gat1_b = (const float*)d_in[p++];
  const float* gat2_w = (const float*)d_in[p++]; const float* gat2_as = (const float*)d_in[p++];
  const float* gat2_ad = (const float*)d_in[p++]; const float* gat2_b = (const float*)d_in[p++];
  const float* gat3_w = (const float*)d_in[p++]; const float* gat3_as = (const float*)d_in[p++];
  const float* gat3_ad = (const float*)d_in[p++]; const float* gat3_b = (const float*)d_in[p++];
  const float* gat4_w = (const float*)d_in[p++]; const float* gat4_as = (const float*)d_in[p++];
  const float* gat4_ad = (const float*)d_in[p++]; const float* gat4_b = (const float*)d_in[p++];
  const float* ep_w1 = (const float*)d_in[p++]; const float* ep_b1 = (const float*)d_in[p++];
  const float* ep_w2 = (const float*)d_in[p++]; const float* ep_b2 = (const float*)d_in[p++];
  const float* ep_w3 = (const float*)d_in[p++]; const float* ep_b3 = (const float*)d_in[p++];

  // ---- workspace bump allocator -------------------------------------------
  char* base = (char*)d_ws;
  size_t off = 0;
  auto alloc = [&](size_t bytes) -> void* {
    void* q = base + off;
    off += (bytes + 255) & ~(size_t)255;
    return q;
  };

  auto packA = [&](const float* in, unsigned short* outp, int rows, int cols,
                   int ldin, int Mp, int Kp) {
    long long tot = (long long)Mp * Kp;
    pack_a_bf16<<<(unsigned)((tot + 255) / 256), 256, 0, stream>>>(in, outp, rows, cols, ldin, Mp, Kp);
  };
  // weight [K,N] -> packed K-major [Np,Kp] bf16, converted once
  auto wslot = [&](const float* w, int K, int Nn) -> unsigned short* {
    int Np = rup(Nn, 64), Kp = rup(K, 32);
    unsigned short* q = (unsigned short*)alloc(sizeof(unsigned short) * (size_t)Np * Kp);
    long long tot = (long long)Np * Kp;
    pack_bT_bf16<<<(unsigned)((tot + 255) / 256), 256, 0, stream>>>(w, q, K, Nn, Np, Kp);
    return q;
  };

  unsigned short* w_nn1_1 = wslot(nn1_w1, 16, 32);
  unsigned short* w_nn1_2 = wslot(nn1_w2, 32, 64);
  unsigned short* w_nn1_3 = wslot(nn1_w3, 64, 32);
  unsigned short* w_gcn1  = wslot(gcn1_w, 32, 64);
  unsigned short* w_gcn2  = wslot(gcn2_w, 64, 128);
  unsigned short* w_gcn3  = wslot(gcn3_w, 128, 256);
  unsigned short* w_gcn4  = wslot(gcn4_w, 256, 256);
  unsigned short* w_gat1  = wslot(gat1_w, 64, 320);
  unsigned short* w_gat2  = wslot(gat2_w, 128, 640);
  unsigned short* w_gat3  = wslot(gat3_w, 256, 1280);
  unsigned short* w_gat4  = wslot(gat4_w, 256, 2560);
  unsigned short* w_ep1   = wslot(ep_w1, 1409, 2816);   // Kp = 1440
  unsigned short* w_ep2   = wslot(ep_w2, 2816, 1408);
  unsigned short* w_ep3   = wslot(ep_w3, 1408, 1);      // Np = 64

  unsigned short* stageA = (unsigned short*)alloc((size_t)CH * 2816 * sizeof(unsigned short));
  float* xwbuf  = (float*)alloc((size_t)N * 2560 * sizeof(float));
  float* t0     = (float*)alloc((size_t)N * 64 * sizeof(float));
  float* t1     = (float*)alloc((size_t)N * 64 * sizeof(float));
  float* hbuf   = (float*)alloc((size_t)N * 32 * sizeof(float));
  float* gtmp   = (float*)alloc((size_t)N * 256 * sizeof(float));
  float* es     = (float*)alloc((size_t)N * 10 * sizeof(float));
  float* ed     = (float*)alloc((size_t)N * 10 * sizeof(float));
  float* alphaB = (float*)alloc((size_t)N * 21 * 10 * sizeof(float));
  float* xf     = (float*)alloc((size_t)N * 704 * sizeof(float));
  float* sq     = (float*)alloc((size_t)N * sizeof(float));
  int*   src1   = (int*)alloc((size_t)N * K1 * sizeof(int));
  int*   src2   = (int*)alloc((size_t)N * K2 * sizeof(int));
  // union region: kNN distance block OR edge-MLP chunk activations
  size_t unionFloats = (size_t)CH * (1409 + 2816 + 1408);  // 23.07M > 2048*8192
  float* ubase = (float*)alloc(unionFloats * sizeof(float));
  float* Dblk  = ubase;
  float* feats = ubase;
  float* hh    = ubase + (size_t)CH * 1409;
  float* h2    = hh + (size_t)CH * 2816;

  auto gemm = [&](const unsigned short* A, const unsigned short* B, const float* bias,
                  float* C, int M, int Nn, int Kp, int ldc, int act) {
    dim3 g((Nn + 63) / 64, (M + 127) / 128);
    wmma_gemm_bf16<<<g, 128, 0, stream>>>(A, B, bias, C, M, Nn, Kp, ldc, act);
  };

  auto knn = [&](const float* feat, int ldf, int dim, int k, int* srcOut) {
    int Kp = rup(dim, 32);
    packA(feat, stageA, N, dim, ldf, N, Kp);
    sqnorm_kernel<<<(N + 255) / 256, 256, 0, stream>>>(feat, sq, N, dim, ldf);
    const int RB = 2048;
    for (int r0 = 0; r0 < N; r0 += RB) {
      // Gram: B = A packed (K-major packed A == B panel for A@A^T)
      gemm(stageA + (size_t)r0 * Kp, stageA, nullptr, Dblk, RB, N, Kp, N, 0);
      long long tot = (long long)RB * N;
      dist_finish_kernel<<<(unsigned)((tot + 255) / 256), 256, 0, stream>>>(Dblk, sq, r0, RB, N);
      topk_kernel<<<RB, 256, 0, stream>>>(Dblk, r0, N, k, srcOut);
    }
  };

  auto gcn = [&](const float* xin, int ldin, int fin, int fout,
                 const unsigned short* wbf, const float* b, const int* src, int k,
                 float* outp) {
    int Kp = rup(fin, 32);
    packA(xin, stageA, N, fin, ldin, N, Kp);
    gemm(stageA, wbf, nullptr, xwbuf, N, fout, Kp, fout, 0);
    long long tot = (long long)N * fout;
    gcn_agg_kernel<<<(unsigned)((tot + 255) / 256), 256, 0, stream>>>(
        xwbuf, src, b, outp, fout, N, fout, k);
  };

  auto gat = [&](const float* xin, int ldin, int fin, int h, int c,
                 const unsigned short* wbf, const float* as, const float* ad,
                 const float* b, const int* src, int k, float* outp, int ldout) {
    int Kp = rup(fin, 32);
    packA(xin, stageA, N, fin, ldin, N, Kp);
    gemm(stageA, wbf, nullptr, xwbuf, N, h * c, Kp, h * c, 0);
    gat_scores_kernel<<<(N * h + 255) / 256, 256, 0, stream>>>(xwbuf, as, ad, es, ed, N, h, c);
    gat_alpha_kernel<<<(N * h + 255) / 256, 256, 0, stream>>>(es, ed, src, alphaB, N, h, k);
    long long tot = (long long)N * c;
    gat_out_kernel<<<(unsigned)((tot + 255) / 256), 256, 0, stream>>>(
        xwbuf, alphaB, src, b, outp, ldout, N, h, c, k);
  };

  // ---- pipeline ------------------------------------------------------------
  knn(x, 16, 4, K1, src1);                       // graph 1 on x[:, :4]

  // nn1 MLP: [N,16]->[N,32]->[N,64]->[N,32] (relu only on last)
  packA(x, stageA, N, 16, 16, N, 32);
  gemm(stageA, w_nn1_1, nn1_b1, t0, N, 32, 32, 32, 0);
  packA(t0, stageA, N, 32, 32, N, 32);
  gemm(stageA, w_nn1_2, nn1_b2, t1, N, 64, 32, 64, 0);
  packA(t1, stageA, N, 64, 64, N, 64);
  gemm(stageA, w_nn1_3, nn1_b3, hbuf, N, 32, 64, 32, 1);

  gcn(hbuf, 32, 32, 64, w_gcn1, gcn1_b, src1, K1, gtmp);
  gat(gtmp, 64, 64, 5, 64, w_gat1, gat1_as, gat1_ad, gat1_b, src1, K1, xf + 0, 704);
  gcn(xf + 0, 704, 64, 128, w_gcn2, gcn2_b, src1, K1, gtmp);
  gat(gtmp, 128, 128, 5, 128, w_gat2, gat2_as, gat2_ad, gat2_b, src1, K1, xf + 64, 704);

  knn(xf + 64, 704, 128, K2, src2);              // graph 2 on x2

  gcn(xf + 64, 704, 128, 256, w_gcn3, gcn3_b, src2, K2, gtmp);
  gat(gtmp, 256, 256, 5, 256, w_gat3, gat3_as, gat3_ad, gat3_b, src2, K2, xf + 192, 704);
  gcn(xf + 192, 704, 256, 256, w_gcn4, gcn4_b, src2, K2, gtmp);
  gat(gtmp, 256, 256, 10, 256, w_gat4, gat4_as, gat4_ad, gat4_b, src2, K2, xf + 448, 704);

  float* outp = (float*)d_out;
  labels_kernel<<<(E2 + 255) / 256, 256, 0, stream>>>(y, src2, outp + E2, E2, K2);

  // edge-predictor MLP, chunked (dominant compute: branch-free bf16 WMMA)
  for (int e0 = 0; e0 < E2; e0 += CH) {
    long long tot = (long long)CH * 1408;
    edge_feat_kernel<<<(unsigned)((tot + 255) / 256), 256, 0, stream>>>(xf, src2, feats, e0, CH, K2);
    edge_weight_kernel<<<(CH + 255) / 256, 256, 0, stream>>>(xf, src2, feats, e0, CH, K2);
    packA(feats, stageA, CH, 1409, 1409, CH, 1440);
    gemm(stageA, w_ep1, ep_b1, hh, CH, 2816, 1440, 2816, 1);
    packA(hh, stageA, CH, 2816, 2816, CH, 2816);
    gemm(stageA, w_ep2, ep_b2, h2, CH, 1408, 2816, 1408, 1);
    packA(h2, stageA, CH, 1408, 1408, CH, 1408);
    gemm(stageA, w_ep3, ep_b3, outp + e0, CH, 1, 1408, 1, 2);  // sigmoid
  }
}
